// SEWResidualBlock_4226247819616
// MI455X (gfx1250) — matrix operand order repaired and checked by reading, not compiled, not run on hardware
//
#include <hip/hip_runtime.h>
#include <hip/hip_fp16.h>

// ---------------------------------------------------------------------------
// SEW residual block for MI455X (gfx1250, wave32, WMMA + Tensor Data Mover).
//   conv1 3x3 s2 p1 (128->256) + BN1 -> LIF(T=8) -> conv2 3x3 s1 p1 + BN2
//   + (1x1 s2 downsample + BNd)  -> LIF(T=8)
// GEMMs: v_wmma_f32_16x16x32_f16 (f16 A/B exact for 0/1 spike activations).
// B (weight) tiles staged by TDM tensor_load_to_lds, double buffered,
// synchronized with s_wait_tensorcnt. A tiles software-pipelined via regs.
// ---------------------------------------------------------------------------

typedef __attribute__((ext_vector_type(16))) _Float16 v16h;
typedef __attribute__((ext_vector_type(8)))  float    v8f;
typedef unsigned int tdm4u __attribute__((ext_vector_type(4)));
typedef int          tdm8i __attribute__((ext_vector_type(8)));
typedef int          tdm4i __attribute__((ext_vector_type(4)));

union FragAB { uint4 q[2]; v16h v; };
union FragC  { v8f  v;  float f[8]; };

// Problem constants
#define BT    128          // B*T
#define BATCH 16
#define TT    8
#define CIN1  128
#define COUT  256
#define HIN   56
#define HOUT  28
#define HWO   784          // 28*28
#define NPIX  (BT * HWO)   // 100352 GEMM N-pixels

// ---------------------------------------------------------------------------
// TDM: load one 32(row) x 128(cout) f16 weight tile into LDS.
// D# group0/group1 packed per CDNA5 ISA 8.3/8.4:
//   g0: count=1 | lds_addr | global_addr(57b) | type=2
//   g1: data_size=1(2B), tensor_dim0=COUT, tensor_dim1=rows,
//       tile_dim0=128, tile_dim1=32, tensor_dim0_stride=COUT
// ---------------------------------------------------------------------------
__device__ __forceinline__ void tdm_load_b_tile(unsigned lds_addr,
                                                unsigned long long gaddr,
                                                int total_rows) {
#if __has_builtin(__builtin_amdgcn_tensor_load_to_lds)
  tdm4u g0;
  g0.x = 1u;                                            // count=1 (user D#)
  g0.y = lds_addr;                                      // LDS byte address
  g0.z = (unsigned)(gaddr & 0xFFFFFFFFull);             // global_addr[31:0]
  g0.w = (unsigned)((gaddr >> 32) & 0x01FFFFFFull)      // global_addr[56:32]
         | (2u << 30);                                  // type=2 ("image")
  tdm8i g1;
  g1[0] = 0x00010000;                                   // data_size=1 (2B)
  g1[1] = (int)(((unsigned)COUT & 0xFFFFu) << 16);      // tensor_dim0[15:0]
  g1[2] = (int)((((unsigned)total_rows & 0xFFFFu) << 16)
                | (((unsigned)COUT >> 16) & 0xFFFFu));  // dim1 lo | dim0 hi
  g1[3] = (int)((128u << 16)                            // tile_dim0 = 128
                | (((unsigned)total_rows >> 16) & 0xFFFFu));
  g1[4] = 32;                                           // tile_dim1 = 32
  g1[5] = COUT;                                         // dim0_stride[31:0]
  g1[6] = 0;
  g1[7] = 0;
  tdm4i gz = {0, 0, 0, 0};
#if __clang_major__ >= 23
  tdm8i gz8 = {0, 0, 0, 0, 0, 0, 0, 0};
  __builtin_amdgcn_tensor_load_to_lds(g0, g1, gz, gz, gz8, 0);
#else
  __builtin_amdgcn_tensor_load_to_lds(g0, g1, gz, gz, 0);
#endif
#endif
}

// ---------------------------------------------------------------------------
// x_spike f32 NCHW [BT][128][56][56] -> f16 NHWC [BT][56][56][128]
// ---------------------------------------------------------------------------
__global__ void k_convert_x(const float* __restrict__ x, __half* __restrict__ xh) {
  size_t i = (size_t)blockIdx.x * 256 + threadIdx.x;
  const size_t total = (size_t)BT * CIN1 * HIN * HIN;
  if (i >= total) return;
  int c = (int)(i & (CIN1 - 1));
  size_t t = i >> 7;
  int w = (int)(t % HIN); t /= HIN;
  int h = (int)(t % HIN);
  int n = (int)(t / HIN);
  xh[i] = __float2half(x[(((size_t)n * CIN1 + c) * HIN + h) * HIN + w]);
}

// ---------------------------------------------------------------------------
// Weight repack: OIHW f32 [Cout][Cin][K*K] -> f16 [tap][Cin][Cout]
// ---------------------------------------------------------------------------
__global__ void k_pack_w(const float* __restrict__ w, __half* __restrict__ wp,
                         int Cout, int Cin, int KK) {
  size_t i = (size_t)blockIdx.x * 256 + threadIdx.x;
  size_t total = (size_t)Cout * Cin * KK;
  if (i >= total) return;
  int co  = (int)(i % Cout);
  size_t r = i / Cout;
  int ci  = (int)(r % Cin);
  int tap = (int)(r / Cin);
  wp[i] = __float2half(w[((size_t)co * Cin + ci) * KK + tap]);
}

// ---------------------------------------------------------------------------
// Implicit-GEMM conv + BN epilogue.
//   A (M=pixels) : NHWC f16 activations  [BT][Hin][Win][CIN]
//   B (N=couts)  : packed f16 weights    [taps][CIN][COUT]  (TDM-staged)
//   D            : NCHW f32              [BT][256][28][28]
// Block 256 thr = 8 waves (4M x 2N). Tile M=128 pix, N=128 couts, K=32/step.
// Pipeline per step s:
//   s_wait_tensorcnt 0 ; barrier        (B[s] landed, everyone done with s-1)
//   commit A regs(s) -> lA ; wave0 issues TDM for B[s+1] ; barrier
//   global-load A regs(s+1)             (latency hides under WMMAs)
//   8 x v_wmma on lA x lB[s&1]
// ---------------------------------------------------------------------------
template<int CIN, int KDIM, int STRIDE, int PAD, bool ADD_ID>
__global__ void __launch_bounds__(256)
k_conv_wmma(const __half* __restrict__ Aact,
            const __half* __restrict__ Wp,
            float* __restrict__ out,
            const float* __restrict__ bn_g, const float* __restrict__ bn_b,
            const float* __restrict__ bn_m, const float* __restrict__ bn_v,
            int Hin, int Win) {
  constexpr int CHUNKS = CIN / 32;
  constexpr int STEPS  = KDIM * KDIM * CHUNKS;

  __shared__ __align__(16) _Float16 lA[128 * 32];       // [pixel][K]     8 KB
  __shared__ __align__(16) _Float16 lB[2][32 * 128];    // [K][cout] x2  16 KB

  const int tid      = threadIdx.x;
  const int pixBase  = blockIdx.x * 128;
  const int coutBase = blockIdx.y * 128;

  // staging-thread pixel decomposition (pixel pi = tid>>1, fixed all kernel)
  const int spi   = tid >> 1;
  const int shalf = (tid & 1) << 4;
  const int sp    = pixBase + spi;
  const int sbt   = sp / HWO;
  const int shw   = sp - sbt * HWO;
  const int soh   = shw / HOUT;
  const int sow   = shw - soh * HOUT;

  const int wave = tid >> 5;
  const int lane = tid & 31;
  const int wm   = wave >> 1;            // 0..3 : M offset wm*32
  const int wn   = wave & 1;             // 0..1 : N offset wn*64

  // A-tile gather for step s (handles 3x3 halo + zero padding)
  auto loadA = [&](int s, uint4& q0, uint4& q1) {
    const int tap = s / CHUNKS;
    const int c0  = (s - tap * CHUNKS) * 32;
    const int ky  = tap / KDIM;
    const int kx  = tap - ky * KDIM;
    const int ih  = soh * STRIDE + ky - PAD;
    const int iw  = sow * STRIDE + kx - PAD;
    const uint4 z = {0, 0, 0, 0};
    q0 = z; q1 = z;
    if (((unsigned)ih < (unsigned)Hin) && ((unsigned)iw < (unsigned)Win)) {
      const uint4* src = (const uint4*)(Aact +
          (((size_t)sbt * Hin + ih) * Win + iw) * CIN + c0 + shalf);
      q0 = src[0];
      q1 = src[1];
    }
  };

  const unsigned long long wTileBase =
      (unsigned long long)(uintptr_t)(Wp + coutBase);        // bytes
  const unsigned ldsB0 = (unsigned)(uintptr_t)(&lB[0][0]);
  const unsigned ldsB1 = (unsigned)(uintptr_t)(&lB[1][0]);

  FragC acc[2][4];
#pragma unroll
  for (int mt = 0; mt < 2; ++mt)
#pragma unroll
    for (int nt = 0; nt < 4; ++nt)
      acc[mt][nt].v = (v8f){};

  // prologue: TDM B[0] in flight, A[0] in registers
  uint4 a0, a1;
  loadA(0, a0, a1);
  if (wave == 0)
    tdm_load_b_tile(ldsB0, wTileBase, STEPS * 32);

  for (int s = 0; s < STEPS; ++s) {
    __builtin_amdgcn_s_wait_tensorcnt(0);   // B[s] landed (wave0; no-op rest)
    __syncthreads();                        // visible to all; s-1 compute done

    // commit A[s] registers to LDS
    *(uint4*)&lA[spi * 32 + shalf]     = a0;
    *(uint4*)&lA[spi * 32 + shalf + 8] = a1;

    // launch DMA for next weight tile into the other buffer
    if (wave == 0 && s + 1 < STEPS)
      tdm_load_b_tile(((s + 1) & 1) ? ldsB1 : ldsB0,
                      wTileBase + (unsigned long long)(s + 1) * 32 * COUT *
                                      sizeof(__half),
                      STEPS * 32);
    __syncthreads();                        // lA ready

    // prefetch A[s+1] into registers; waitcnt lands after the WMMA block
    if (s + 1 < STEPS) loadA(s + 1, a0, a1);

    // fragment loads matching CDNA5 wave32 WMMA VGPR layouts
    // A 16x32 f16: lanes 0-15 row M=lane, K {0-7,16-23};
    //              lanes 16-31 row M=lane-16, K {8-15,24-31}
    const _Float16* lBc = &lB[s & 1][0];
    const int aRow = lane & 15;
    const int kOff = (lane >> 4) << 3;
    FragAB afrag[2];
#pragma unroll
    for (int mt = 0; mt < 2; ++mt) {
      const _Float16* p = &lA[(wm * 32 + mt * 16 + aRow) * 32];
      afrag[mt].q[0] = *(const uint4*)(p + kOff);
      afrag[mt].q[1] = *(const uint4*)(p + 16 + kOff);
    }
    // B 32x16 f16: N col = lane&15; K {0-7,16-23} (lanes<16) / {8-15,24-31}
    FragAB bfrag[4];
#pragma unroll
    for (int nt = 0; nt < 4; ++nt) {
      const _Float16* p = lBc + wn * 64 + nt * 16 + (lane & 15);
#pragma unroll
      for (int i = 0; i < 8; ++i) {
        bfrag[nt].v[i]     = p[(kOff + i) * 128];
        bfrag[nt].v[8 + i] = p[(16 + kOff + i) * 128];
      }
    }

#pragma unroll
    for (int mt = 0; mt < 2; ++mt)
#pragma unroll
      for (int nt = 0; nt < 4; ++nt)
        acc[mt][nt].v = __builtin_amdgcn_wmma_f32_16x16x32_f16(
            false, afrag[mt].v, false, bfrag[nt].v,
            (short)0, acc[mt][nt].v, false, false);
  }

  // epilogue: fused inference BN, optional identity add (in place)
  // D 16x16 f32: N col = lane&15; VGPR r -> M row r (lanes<16) / r+8
  const int nLane = lane & 15;
  const int rOff  = (lane >> 4) << 3;
  float sc[4], sh[4];
#pragma unroll
  for (int nt = 0; nt < 4; ++nt) {
    const int c = coutBase + wn * 64 + nt * 16 + nLane;
    const float s = bn_g[c] * rsqrtf(bn_v[c] + 1e-5f);
    sc[nt] = s;
    sh[nt] = bn_b[c] - bn_m[c] * s;
  }
#pragma unroll
  for (int mt = 0; mt < 2; ++mt) {
#pragma unroll
    for (int r = 0; r < 8; ++r) {
      const int p  = pixBase + wm * 32 + mt * 16 + rOff + r;
      const int bt = p / HWO;
      const int hw = p - bt * HWO;
#pragma unroll
      for (int nt = 0; nt < 4; ++nt) {
        const int c = coutBase + wn * 64 + nt * 16 + nLane;
        const size_t o = ((size_t)bt * COUT + c) * HWO + hw;
        float val = acc[mt][nt].f[r] * sc[nt] + sh[nt];
        if (ADD_ID) val += out[o];
        out[o] = val;
      }
    }
  }
}

// ---------------------------------------------------------------------------
// LIF scan #1: pre-LIF conv1+BN (f32 NCHW) -> spikes f16 NHWC (conv2 A input)
// v' = (v+x)/2 ; s = (v' > 1) ; v' *= (1-s)
// ---------------------------------------------------------------------------
__global__ void k_lif1(const float* __restrict__ y1, __half* __restrict__ s1) {
  size_t i = (size_t)blockIdx.x * 256 + threadIdx.x;
  const size_t total = (size_t)BATCH * COUT * HWO;
  if (i >= total) return;
  const int hw = (int)(i % HWO);
  size_t t1   = i / HWO;
  const int c = (int)(t1 % COUT);
  const int b = (int)(t1 / COUT);
  float vmem = 0.f;
#pragma unroll
  for (int t = 0; t < TT; ++t) {
    const int bt = b * TT + t;
    const float x = y1[((size_t)bt * COUT + c) * HWO + hw];
    vmem = 0.5f * (vmem + x);
    const float s = (vmem > 1.0f) ? 1.0f : 0.0f;
    s1[((size_t)bt * HWO + hw) * COUT + c] = __float2half(s);
    vmem *= (1.0f - s);
  }
}

// ---------------------------------------------------------------------------
// LIF scan #2: in place on d_out (f32 NCHW); one thread owns each (b,c,h,w)
// ---------------------------------------------------------------------------
__global__ void k_lif2(float* __restrict__ out) {
  size_t i = (size_t)blockIdx.x * 256 + threadIdx.x;
  const size_t total = (size_t)BATCH * COUT * HWO;
  if (i >= total) return;
  const int hw = (int)(i % HWO);
  size_t t1   = i / HWO;
  const int c = (int)(t1 % COUT);
  const int b = (int)(t1 / COUT);
  float vmem = 0.f;
#pragma unroll
  for (int t = 0; t < TT; ++t) {
    const size_t o = ((size_t)(b * TT + t) * COUT + c) * HWO + hw;
    const float x = out[o];
    vmem = 0.5f * (vmem + x);
    const float s = (vmem > 1.0f) ? 1.0f : 0.0f;
    out[o] = s;
    vmem *= (1.0f - s);
  }
}

// ---------------------------------------------------------------------------
// Host-side orchestration (graph-capture safe: only kernel launches on stream)
// ---------------------------------------------------------------------------
extern "C" void kernel_launch(void* const* d_in, const int* in_sizes, int n_in,
                              void* d_out, int out_size, void* d_ws, size_t ws_size,
                              hipStream_t stream) {
  const float* x   = (const float*)d_in[0];   // [16,8,128,56,56]
  const float* w1  = (const float*)d_in[1];   // [256,128,3,3]
  const float* w2  = (const float*)d_in[2];   // [256,256,3,3]
  const float* wd  = (const float*)d_in[3];   // [256,128,1,1]
  const float* n1g = (const float*)d_in[4],  *n1b = (const float*)d_in[5];
  const float* n1m = (const float*)d_in[6],  *n1v = (const float*)d_in[7];
  const float* n2g = (const float*)d_in[8],  *n2b = (const float*)d_in[9];
  const float* n2m = (const float*)d_in[10], *n2v = (const float*)d_in[11];
  const float* ndg = (const float*)d_in[12], *ndb = (const float*)d_in[13];
  const float* ndm = (const float*)d_in[14], *ndv = (const float*)d_in[15];
  float* out = (float*)d_out;                 // [16,8,256,28,28] f32

  // Workspace layout (~247 MiB)
  const size_t XH_BYTES = (size_t)BT * CIN1 * HIN * HIN * sizeof(__half);
  const size_t Y1_BYTES = (size_t)BT * COUT * HWO * sizeof(float);
  const size_t S1_BYTES = (size_t)BT * COUT * HWO * sizeof(__half);
  const size_t W1_BYTES = (size_t)9 * CIN1 * COUT * sizeof(__half);
  const size_t W2_BYTES = (size_t)9 * COUT * COUT * sizeof(__half);
  char* ws = (char*)d_ws;
  __half* xh  = (__half*)(ws);
  float*  y1  = (float*) (ws + XH_BYTES);
  __half* s1  = (__half*)(ws + XH_BYTES + Y1_BYTES);
  __half* w1p = (__half*)(ws + XH_BYTES + Y1_BYTES + S1_BYTES);
  __half* w2p = (__half*)(ws + XH_BYTES + Y1_BYTES + S1_BYTES + W1_BYTES);
  __half* wdp = (__half*)(ws + XH_BYTES + Y1_BYTES + S1_BYTES + W1_BYTES + W2_BYTES);

  // 1) format conversions / weight repack
  {
    const size_t nx = (size_t)BT * CIN1 * HIN * HIN;
    k_convert_x<<<dim3((unsigned)((nx + 255) / 256)), 256, 0, stream>>>(x, xh);
    k_pack_w<<<dim3((COUT * CIN1 * 9 + 255) / 256), 256, 0, stream>>>(w1, w1p, COUT, CIN1, 9);
    k_pack_w<<<dim3((COUT * COUT * 9 + 255) / 256), 256, 0, stream>>>(w2, w2p, COUT, COUT, 9);
    k_pack_w<<<dim3((COUT * CIN1 * 1 + 255) / 256), 256, 0, stream>>>(wd, wdp, COUT, CIN1, 1);
  }

  const dim3 ggrid(NPIX / 128, COUT / 128);   // (784, 2)

  // 2) downsample 1x1 s2 + BNd -> d_out (identity accumulator)
  k_conv_wmma<CIN1, 1, 2, 0, false><<<ggrid, 256, 0, stream>>>(
      xh, wdp, out, ndg, ndb, ndm, ndv, HIN, HIN);

  // 3) conv1 3x3 s2 p1 + BN1 -> y1 (pre-LIF, f32 NCHW)
  k_conv_wmma<CIN1, 3, 2, 1, false><<<ggrid, 256, 0, stream>>>(
      xh, w1p, y1, n1g, n1b, n1m, n1v, HIN, HIN);

  // 4) LIF over T -> spikes f16 NHWC
  k_lif1<<<dim3((BATCH * COUT * HWO) / 256), 256, 0, stream>>>(y1, s1);

  // 5) conv2 3x3 s1 p1 + BN2 + identity (in d_out) -> d_out
  k_conv_wmma<COUT, 3, 1, 1, true><<<ggrid, 256, 0, stream>>>(
      s1, w2p, out, n2g, n2b, n2m, n2v, HOUT, HOUT);

  // 6) final LIF in place on d_out
  k_lif2<<<dim3((BATCH * COUT * HWO) / 256), 256, 0, stream>>>(out);
}
